// Int4WeightInt8ActLinear_7980049236644
// MI455X (gfx1250) — 8-bit matrix operand order repaired and checked by reading, compiled, not run on hardware
//
#include <hip/hip_runtime.h>
#include <stdint.h>

typedef __attribute__((ext_vector_type(8))) int v8i;

#define B_DIM 32768
#define IN_F  1024
#define OUT_F 1024
#define KT    (IN_F / 64)
#define MT    (B_DIM / 16)
#define NT    (OUT_F / 16)

#define QA_BYTES ((size_t)MT * KT * 1024)

__global__ void __launch_bounds__(256)
quant_kernel(const float* __restrict__ x, uint32_t* __restrict__ qA) {
    uint32_t t    = blockIdx.x * 256u + threadIdx.x;
    uint32_t tile = t >> 8;
    uint32_t slot = t & 255u;
    uint32_t lane = slot >> 3;
    uint32_t v    = slot & 7u;
    uint32_t mt   = tile >> 4;
    uint32_t kt   = tile & 15u;
    uint32_t m    = mt * 16u + (lane & 15u);
    uint32_t hi   = lane >> 4;
    uint32_t k    = kt * 64u + ((v >> 2) << 5) + ((v & 3u) << 2)
                  + (hi << 4);

    const float4 xv = *(const float4*)(x + (size_t)m * IN_F + k);
    float f0 = fminf(127.f, fmaxf(-127.f, rintf(xv.x * 20.0f)));
    float f1 = fminf(127.f, fmaxf(-127.f, rintf(xv.y * 20.0f)));
    float f2 = fminf(127.f, fmaxf(-127.f, rintf(xv.z * 20.0f)));
    float f3 = fminf(127.f, fmaxf(-127.f, rintf(xv.w * 20.0f)));
    uint32_t d = ((uint32_t)(int)f0 & 0xFFu)
               | (((uint32_t)(int)f1 & 0xFFu) << 8)
               | (((uint32_t)(int)f2 & 0xFFu) << 16)
               | (((uint32_t)(int)f3 & 0xFFu) << 24);
    qA[t] = d;
}

__global__ void __launch_bounds__(256)
unpack_kernel(const uint8_t* __restrict__ wp, uint8_t* __restrict__ wB) {
    uint32_t t    = blockIdx.x * 256u + threadIdx.x;
    uint32_t lane = t & 31u;
    uint32_t nt   = (t >> 5) & 63u;
    uint32_t kt   = t >> 11;
    uint32_t n    = nt * 16u + (lane & 15u);
    uint32_t khi  = (lane >> 4) << 4;

    uint32_t* dst = (uint32_t*)(wB + ((size_t)(kt * 64u + nt) * 1024u + lane * 32u));
#pragma unroll
    for (int g = 0; g < 2; ++g) {
        uint64_t p = *(const uint64_t*)(wp + (size_t)n * (IN_F / 2)
                                        + kt * 32u + (uint32_t)g * 16u + (khi >> 1));
        uint32_t d[4] = {0u, 0u, 0u, 0u};
#pragma unroll
        for (int j = 0; j < 8; ++j) {
            uint32_t byte = (uint32_t)(p >> (8 * j)) & 0xFFu;
            int lo = ((int)(byte << 28)) >> 28;
            int hn = ((int)(byte << 24)) >> 28;
            uint32_t pos = 2u * (uint32_t)j;
            d[pos >> 2] |= ((uint32_t)lo & 0xFFu) << ((pos & 3u) * 8u);
            pos++;
            d[pos >> 2] |= ((uint32_t)hn & 0xFFu) << ((pos & 3u) * 8u);
        }
        dst[g * 4 + 0] = d[0]; dst[g * 4 + 1] = d[1];
        dst[g * 4 + 2] = d[2]; dst[g * 4 + 3] = d[3];
    }
}

__global__ void __launch_bounds__(128)
gemm_kernel(const uint8_t* __restrict__ qA, const uint8_t* __restrict__ wB,
            const float* __restrict__ bias, _Float16* __restrict__ out) {
    const uint32_t lane = threadIdx.x & 31u;
    const uint32_t wave = threadIdx.x >> 5;
    const uint32_t mt0  = blockIdx.y * 8u + wave * 2u;
    const uint32_t nt0  = blockIdx.x * 4u;

    const uint8_t* a0p = qA + (size_t)(mt0 + 0) * 16u * 1024u + lane * 32u;
    const uint8_t* a1p = qA + (size_t)(mt0 + 1) * 16u * 1024u + lane * 32u;
    const uint8_t* bp  = wB + (size_t)nt0 * 1024u + lane * 32u;

    v8i c00 = {}, c01 = {}, c02 = {}, c03 = {};
    v8i c10 = {}, c11 = {}, c12 = {}, c13 = {};

#pragma unroll 4
    for (int kt = 0; kt < KT; ++kt) {
        v8i a0 = *(const v8i*)(a0p + (size_t)kt * 1024u);
        v8i a1 = *(const v8i*)(a1p + (size_t)kt * 1024u);
        v8i b0 = *(const v8i*)(bp + (size_t)kt * 64u * 1024u);
        v8i b1 = *(const v8i*)(bp + (size_t)kt * 64u * 1024u + 1024u);
        v8i b2 = *(const v8i*)(bp + (size_t)kt * 64u * 1024u + 2048u);
        v8i b3 = *(const v8i*)(bp + (size_t)kt * 64u * 1024u + 3072u);
        __builtin_prefetch(a0p + (size_t)(kt + 2) * 1024u, 0, 3);
        __builtin_prefetch(bp + (size_t)(kt + 2) * 64u * 1024u, 0, 3);
        c00 = __builtin_amdgcn_wmma_i32_16x16x64_iu8(true, a0, true, b0, c00, false, false);
        c01 = __builtin_amdgcn_wmma_i32_16x16x64_iu8(true, a0, true, b1, c01, false, false);
        c02 = __builtin_amdgcn_wmma_i32_16x16x64_iu8(true, a0, true, b2, c02, false, false);
        c03 = __builtin_amdgcn_wmma_i32_16x16x64_iu8(true, a0, true, b3, c03, false, false);
        c10 = __builtin_amdgcn_wmma_i32_16x16x64_iu8(true, a1, true, b0, c10, false, false);
        c11 = __builtin_amdgcn_wmma_i32_16x16x64_iu8(true, a1, true, b1, c11, false, false);
        c12 = __builtin_amdgcn_wmma_i32_16x16x64_iu8(true, a1, true, b2, c12, false, false);
        c13 = __builtin_amdgcn_wmma_i32_16x16x64_iu8(true, a1, true, b3, c13, false, false);
    }

    const uint32_t nloc  = lane & 15u;
    const uint32_t mhalf = (lane >> 4) * 8u;
    const float    deq   = 5.0e-4f;
    float bv[4];
#pragma unroll
    for (int j = 0; j < 4; ++j) bv[j] = bias[(nt0 + (uint32_t)j) * 16u + nloc];

    const v8i* cc[2][4] = { { &c00, &c01, &c02, &c03 },
                            { &c10, &c11, &c12, &c13 } };
#pragma unroll
    for (int i = 0; i < 2; ++i) {
#pragma unroll
        for (int j = 0; j < 4; ++j) {
            const v8i c = *cc[i][j];
            size_t col = (size_t)(nt0 + (uint32_t)j) * 16u + nloc;
#pragma unroll
            for (int r = 0; r < 8; ++r) {
                size_t row = (size_t)(mt0 + (uint32_t)i) * 16u + mhalf + (uint32_t)r;
                out[row * OUT_F + col] = (_Float16)((float)c[r] * deq + bv[j]);
            }
        }
    }
}

extern "C" void kernel_launch(void* const* d_in, const int* in_sizes, int n_in,
                              void* d_out, int out_size, void* d_ws, size_t ws_size,
                              hipStream_t stream) {
    const float*   x    = (const float*)d_in[0];
    const uint8_t* wp   = (const uint8_t*)d_in[1];
    const float*   bias = (const float*)d_in[2];
    _Float16*      out  = (_Float16*)d_out;

    uint8_t* qA = (uint8_t*)d_ws;
    uint8_t* wB = qA + QA_BYTES;

    quant_kernel<<<MT * KT, 256, 0, stream>>>(x, (uint32_t*)qA);
    unpack_kernel<<<(KT * NT * 32) / 256, 256, 0, stream>>>(wp, wB);
    gemm_kernel<<<dim3(OUT_F / 64, B_DIM / 128), 128, 0, stream>>>(qA, wB, bias, out);
}
